// VQVAE_85358180041008
// MI455X (gfx1250) — matrix operand order repaired and checked by reading, not compiled, not run on hardware
//
#include <hip/hip_runtime.h>
#include <math.h>

typedef __attribute__((ext_vector_type(16))) _Float16 v16h;
typedef __attribute__((ext_vector_type(8)))  float    v8f;

// ---- d_out float offsets (x_recon, vq_loss, indices, seg_logits) ----
#define RECON_OFF  0L
#define VQLOSS_OFF 3145728L
#define IDX_OFF    3145729L
#define SEG_OFF    3162113L

// ---- workspace float offsets ----
#define WA   0L          // 8,388,608  (16x32x128x128 / dt2 out / seg up2#2)
#define WB   8388608L    // 4,194,304  (16x64x64x64 / dt1 out / seg up2#1)
#define WC   12582912L   // 4,980,736  (16x19x128x128 seg s3 out)
#define WT0  17563648L   // 2,097,152  z_e
#define WT1  19660800L   // 2,097,152  z_q
#define WT2  21757952L   // 2,097,152  resblock accumulator (h3 / d)
#define WT3  23855104L   // 2,097,152  temp
#define WT4  25952256L   // 2,097,152  temp
#define WCN  28049408L   // 1,024      codebook norms
#define WLB  28050432L   // 1,024      per-block loss partials
#define WS_FLOATS 28051456L

// A-fragment (16x32 f16) per-lane k map: VGPR0-3 hold K 0-7 (+8 for lanes 16-31),
// VGPR4-7 hold K 16-23 (+8 for lanes 16-31); two packed halves per VGPR.
__device__ __forceinline__ int a_kmap(int e, int g){
  int v = e >> 1, lo = e & 1;
  return ((v < 4) ? (2 * v) : (16 + 2 * (v - 4))) + 8 * g + lo;
}

// ---------------- generic conv / lhs-dilated conv as implicit GEMM ----------------
// CIN % 32 == 0. DIL==2 => ConvTranspose2d(k=4,s=2,p=1) as stride-1/pad-2 conv on
// the 2x lhs-dilated input. ACT: 0=none, 1=relu, 2=sigmoid.
// Hout = 1<<hoSh, Wout = 1<<woSh (all spatial dims in this net are powers of two).
template<int CIN, int DIL, int ACT>
__global__ __launch_bounds__(32)
void k_conv_wmma(const float* __restrict__ in, const float* __restrict__ w,
                 const float* __restrict__ bias, float* __restrict__ out,
                 int N, int Hin, int Win, int Cout,
                 int hoSh, int woSh, int ksz, int stride, int pad)
{
  const int lane = threadIdx.x & 31;
  const int g = lane >> 4;
  const int l = lane & 15;
  const int pixBase = blockIdx.x << 4;
  const int coBase  = blockIdx.y << 4;
  const int phw = hoSh + woSh;
  const int npix = N << phw;
  const int hw = Hin * Win;

  int p = pixBase + l;
  bool pvalid = p < npix;
  int pc = pvalid ? p : 0;
  int ox = pc & ((1 << woSh) - 1);
  int oy = (pc >> woSh) & ((1 << hoSh) - 1);
  int nimg = pc >> phw;
  int ty0 = oy * stride - pad;
  int tx0 = ox * stride - pad;

  const int co_a = coBase + l;
  const bool cok = co_a < Cout;
  const int kk2 = ksz * ksz;

  v8f acc = {};
  for (int ky = 0; ky < ksz; ++ky){
    int ta = ty0 + ky;
    int iy; bool yok;
    if (DIL == 2){ yok = (ta >= 0) && ((ta & 1) == 0); iy = ta >> 1; yok = yok && (iy < Hin); }
    else         { yok = (ta >= 0) && (ta < Hin); iy = ta; }
    for (int kx = 0; kx < ksz; ++kx){
      int tb = tx0 + kx;
      int ix; bool xok;
      if (DIL == 2){ xok = (tb >= 0) && ((tb & 1) == 0); ix = tb >> 1; xok = xok && (ix < Win); }
      else         { xok = (tb >= 0) && (tb < Win); ix = tb; }
      bool inb = pvalid && yok && xok;
      const float* ib = in + (nimg * CIN) * hw + (inb ? (iy * Win + ix) : 0);
      const float* wb = w + ((co_a * CIN) * ksz + ky) * ksz + kx;
#pragma unroll
      for (int c0 = 0; c0 < CIN; c0 += 32){
        v16h a, b;
#pragma unroll
        for (int e = 0; e < 16; ++e){
          int ka = c0 + a_kmap(e, g);
          float av = cok ? wb[ka * kk2] : 0.f;
          a[e] = (_Float16)av;
          int kb = c0 + a_kmap(e, g);
          float bv = inb ? ib[kb * hw] : 0.f;
          b[e] = (_Float16)bv;
        }
        acc = __builtin_amdgcn_wmma_f32_16x16x32_f16(false, a, false, b, (short)0, acc, false, false);
      }
    }
  }
#pragma unroll
  for (int r = 0; r < 8; ++r){
    int co = coBase + r + (g << 3);
    int p2 = pixBase + l;
    if (co < Cout && p2 < npix){
      float v = acc[r] + bias[co];
      if (ACT == 1) v = fmaxf(v, 0.f);
      else if (ACT == 2) v = 1.f / (1.f + __expf(-v));
      int ox2 = p2 & ((1 << woSh) - 1);
      int oy2 = (p2 >> woSh) & ((1 << hoSh) - 1);
      int n2 = p2 >> phw;
      out[((n2 * Cout + co) << phw) + (oy2 << woSh) + ox2] = v;
    }
  }
}

// ---------------- e1: Cin=3 scalar conv (4x4 s2 p1) + relu ----------------
__global__ void k_conv_e1(const float* __restrict__ x, const float* __restrict__ w,
                          const float* __restrict__ b, float* __restrict__ out)
{
  int i = blockIdx.x * 256 + threadIdx.x;            // 16*32*128*128
  if (i >= 16 * 32 * 128 * 128) return;
  int ox = i & 127; int t = i >> 7; int oy = t & 127; t >>= 7; int co = t & 31; int n = t >> 5;
  float acc = b[co];
  int iy0 = oy * 2 - 1, ix0 = ox * 2 - 1;
  for (int ci = 0; ci < 3; ++ci)
    for (int ky = 0; ky < 4; ++ky){
      int iy = iy0 + ky; if (iy < 0 || iy >= 256) continue;
      for (int kx = 0; kx < 4; ++kx){
        int ix = ix0 + kx; if (ix < 0 || ix >= 256) continue;
        acc += w[((co * 3 + ci) * 4 + ky) * 4 + kx] * x[((n * 3 + ci) * 256 + iy) * 256 + ix];
      }
    }
  out[i] = fmaxf(acc, 0.f);
}

// ---------------- GroupNorm(8 groups) + optional ReLU, fixed 16x128x32x32 ----------------
__global__ void k_gn(const float* __restrict__ in, const float* __restrict__ sc,
                     const float* __restrict__ bi, float* __restrict__ out, int relu)
{
  int blk = blockIdx.x;                 // 0..127 = (n, group)
  int n = blk >> 3, g = blk & 7;
  const float* base = in + ((n * 128 + g * 16) << 10);
  float* obase = out + ((n * 128 + g * 16) << 10);
  int tid = threadIdx.x;
  float s = 0.f, s2 = 0.f;
  for (int i = tid; i < 16384; i += 256){ float v = base[i]; s += v; s2 += v * v; }
  __shared__ float rs[256], rq[256];
  rs[tid] = s; rq[tid] = s2; __syncthreads();
  for (int o = 128; o > 0; o >>= 1){
    if (tid < o){ rs[tid] += rs[tid + o]; rq[tid] += rq[tid + o]; }
    __syncthreads();
  }
  __shared__ float ms, rst;
  if (tid == 0){
    float m = rs[0] * (1.f / 16384.f);
    float var = rq[0] * (1.f / 16384.f) - m * m;
    ms = m; rst = rsqrtf(var + 1e-5f);
  }
  __syncthreads();
  float m = ms, r = rst;
  for (int i = tid; i < 16384; i += 256){
    int c = g * 16 + (i >> 10);
    float v = (base[i] - m) * r * sc[c] + bi[c];
    if (relu) v = fmaxf(v, 0.f);
    obase[i] = v;
  }
}

__global__ void k_add(float* __restrict__ dst, const float* __restrict__ src, int n){
  int i = blockIdx.x * 256 + threadIdx.x;
  if (i < n) dst[i] += src[i];
}

// ---------------- bilinear 2x upsample (align_corners=False, edge clamp), N=16 ----------------
__global__ void k_up2(const float* __restrict__ in, float* __restrict__ out,
                      int C, int Hin, int Win)
{
  long total = 16L * C * 4L * Hin * Win;
  long i = (long)blockIdx.x * 256 + threadIdx.x;
  if (i >= total) return;
  int Wo = 2 * Win, Ho = 2 * Hin;
  int ox = (int)(i % Wo); long t = i / Wo; int oy = (int)(t % Ho); t /= Ho;
  int c = (int)(t % C); int n = (int)(t / C);
  float sy = oy * 0.5f - 0.25f, sx = ox * 0.5f - 0.25f;
  int y0 = (int)floorf(sy); float fy = sy - (float)y0;
  int x0 = (int)floorf(sx); float fx = sx - (float)x0;
  int y0c = y0 < 0 ? 0 : (y0 > Hin - 1 ? Hin - 1 : y0);
  int y1 = y0 + 1; int y1c = y1 < 0 ? 0 : (y1 > Hin - 1 ? Hin - 1 : y1);
  int x0c = x0 < 0 ? 0 : (x0 > Win - 1 ? Win - 1 : x0);
  int x1 = x0 + 1; int x1c = x1 < 0 ? 0 : (x1 > Win - 1 ? Win - 1 : x1);
  const float* p = in + ((long)(n * C + c)) * Hin * Win;
  float v00 = p[y0c * Win + x0c], v01 = p[y0c * Win + x1c];
  float v10 = p[y1c * Win + x0c], v11 = p[y1c * Win + x1c];
  out[i] = (1.f - fy) * ((1.f - fx) * v00 + fx * v01) + fy * ((1.f - fx) * v10 + fx * v11);
}

// ---------------- VQ ----------------
__global__ void k_codenorm(const float* __restrict__ cb, float* __restrict__ cn){
  int k = blockIdx.x * 256 + threadIdx.x;
  if (k >= 1024) return;
  const float* r = cb + k * 128;
  float s = 0.f;
  for (int c = 0; c < 128; ++c){ float v = r[c]; s += v * v; }
  cn[k] = s;
}

// per wave: 16 tokens vs all 1024 codes. dist = |c|^2 - 2 z.c (|z|^2 cancels in argmin).
__global__ __launch_bounds__(32)
void k_vq(const float* __restrict__ ze, const float* __restrict__ cb,
          const float* __restrict__ cnorm, float* __restrict__ zq,
          float* __restrict__ idx_out, float* __restrict__ lossbuf)
{
  const int lane = threadIdx.x & 31;
  const int g = lane >> 4, l = lane & 15;
  const int tokBase = blockIdx.x << 4;

  int token = tokBase + l;
  int txp = token & 31, typ = (token >> 5) & 31, tn = token >> 10;
  const float* zb = ze + (tn * 128) * 1024 + typ * 32 + txp;

  v16h afr[4];
#pragma unroll
  for (int c0 = 0; c0 < 4; ++c0){
#pragma unroll
    for (int e = 0; e < 16; ++e){
      int k = c0 * 32 + a_kmap(e, g);
      afr[c0][e] = (_Float16)zb[k * 1024];
    }
  }
  float minv[8]; int mini[8];
#pragma unroll
  for (int r = 0; r < 8; ++r){ minv[r] = 3.4e38f; mini[r] = 0; }

  for (int cbase = 0; cbase < 1024; cbase += 16){
    const float* crow = cb + (cbase + l) * 128;
    v8f acc = {};
#pragma unroll
    for (int c0 = 0; c0 < 4; ++c0){
      v16h b;
#pragma unroll
      for (int e = 0; e < 16; ++e) b[e] = (_Float16)crow[c0 * 32 + a_kmap(e, g)];
      acc = __builtin_amdgcn_wmma_f32_16x16x32_f16(false, afr[c0], false, b, (short)0, acc, false, false);
    }
    float cn = cnorm[cbase + l];
#pragma unroll
    for (int r = 0; r < 8; ++r){
      float sc = cn - 2.f * acc[r];
      if (sc < minv[r]){ minv[r] = sc; mini[r] = cbase + l; }
    }
  }

  __shared__ float sval[16][16];
  __shared__ int   sidx[16][16];
  __shared__ float sls[16];
#pragma unroll
  for (int r = 0; r < 8; ++r){
    int tl = r + (g << 3);
    sval[tl][l] = minv[r];
    sidx[tl][l] = mini[r];
  }
  __syncthreads();
  if (lane < 16){
    int tl = lane;
    float best = sval[tl][0]; int bi = sidx[tl][0];
#pragma unroll
    for (int j = 1; j < 16; ++j)
      if (sval[tl][j] < best){ best = sval[tl][j]; bi = sidx[tl][j]; }
    int tok2 = tokBase + tl;
    int x2 = tok2 & 31, y2 = (tok2 >> 5) & 31, n2 = tok2 >> 10;
    idx_out[tok2] = (float)bi;
    const float* crow = cb + bi * 128;
    const float* zb2 = ze + (n2 * 128) * 1024 + y2 * 32 + x2;
    float* qb = zq + (n2 * 128) * 1024 + y2 * 32 + x2;
    float ls = 0.f;
    for (int c = 0; c < 128; ++c){
      float q = crow[c];
      qb[c * 1024] = q;
      float d = q - zb2[c * 1024];
      ls += d * d;
    }
    sls[tl] = ls;
  }
  __syncthreads();
  if (lane == 0){
    float s = 0.f;
#pragma unroll
    for (int j = 0; j < 16; ++j) s += sls[j];
    lossbuf[blockIdx.x] = s;
  }
}

__global__ void k_vq_fin(const float* __restrict__ lossbuf, float* __restrict__ out){
  float s = 0.f;
  for (int i = 0; i < 1024; ++i) s += lossbuf[i];
  out[0] = 1.25f * s / (16384.f * 128.f);   // codebook + 0.25*commit, both == MSE in fwd
}

// ---------------- host-side helpers ----------------
static inline int ilog2(int v){ int s = 0; while ((1 << s) < v) ++s; return s; }

static inline void convl(hipStream_t s, const float* in, const float* w, const float* b, float* out,
                         int N, int Cin, int Hin, int Win, int Cout, int Hout, int Wout,
                         int ksz, int stride, int pad, int dil, int act)
{
  int hoSh = ilog2(Hout), woSh = ilog2(Wout);
  int npix = N << (hoSh + woSh);
  dim3 grid((unsigned)((npix + 15) / 16), (unsigned)((Cout + 15) / 16));
#define CONV_CASE(CI, DI, AC)                                                      \
  if (Cin == CI && dil == DI && act == AC){                                        \
    k_conv_wmma<CI, DI, AC><<<grid, 32, 0, s>>>(in, w, b, out, N, Hin, Win, Cout,  \
                                                hoSh, woSh, ksz, stride, pad);     \
    return; }
  CONV_CASE(32, 1, 1)     // e2
  CONV_CASE(64, 1, 1)     // e3, s2
  CONV_CASE(128, 1, 0)    // resblock convs, e4, d1
  CONV_CASE(128, 1, 1)    // s1
  CONV_CASE(32, 1, 0)     // s3
  CONV_CASE(128, 2, 1)    // dt1
  CONV_CASE(64, 2, 1)     // dt2
  CONV_CASE(32, 2, 2)     // dt3 (sigmoid)
#undef CONV_CASE
}

static inline void resblock(hipStream_t s, float* x, float* t1, float* t2,
                            const float* g1s, const float* g1b, const float* c1w, const float* c1b,
                            const float* g2s, const float* g2b, const float* c2w, const float* c2b)
{
  k_gn<<<128, 256, 0, s>>>(x, g1s, g1b, t1, 1);
  convl(s, t1, c1w, c1b, t2, 16, 128, 32, 32, 128, 32, 32, 3, 1, 1, 1, 0);
  k_gn<<<128, 256, 0, s>>>(t2, g2s, g2b, t1, 1);
  convl(s, t1, c2w, c2b, t2, 16, 128, 32, 32, 128, 32, 32, 3, 1, 1, 1, 0);
  k_add<<<(2097152 + 255) / 256, 256, 0, s>>>(x, t2, 2097152);
}

// ---------------- parameter slot mapping ----------------
enum {
  iX = 0,
  iE1W, iE1B, iE2W, iE2B, iE3W, iE3B,
  iERB1_G1S, iERB1_G1B, iERB1_C1W, iERB1_C1B, iERB1_G2S, iERB1_G2B, iERB1_C2W, iERB1_C2B,
  iERB2_G1S, iERB2_G1B, iERB2_C1W, iERB2_C1B, iERB2_G2S, iERB2_G2B, iERB2_C2W, iERB2_C2B,
  iE4W, iE4B, iD1W, iD1B,
  iDRB1_G1S, iDRB1_G1B, iDRB1_C1W, iDRB1_C1B, iDRB1_G2S, iDRB1_G2B, iDRB1_C2W, iDRB1_C2B,
  iDRB2_G1S, iDRB2_G1B, iDRB2_C1W, iDRB2_C1B, iDRB2_G2S, iDRB2_G2B, iDRB2_C2W, iDRB2_C2B,
  iDT1W, iDT1B, iDT2W, iDT2B, iDT3W, iDT3B,
  iS1W, iS1B, iS2W, iS2B, iS3W, iS3B,
  iCODEBOOK, N_T
};

// insertion-order flattening: x first, then params in dict insertion order
static const int INS[N_T] = {
  0,
  1, 2, 3, 4, 5, 6,
  7, 8, 9, 10, 11, 12, 13, 14,
  15, 16, 17, 18, 19, 20, 21, 22,
  23, 24, 25, 26,
  27, 28, 29, 30, 31, 32, 33, 34,
  35, 36, 37, 38, 39, 40, 41, 42,
  43, 44, 45, 46, 47, 48,
  49, 50, 51, 52, 53, 54,
  55
};
// jax tree_leaves (sorted keys) flattening: params (sorted) first, x last
static const int SRT[N_T] = {
  55,
  26, 25, 28, 27, 30, 29,
  38, 37, 34, 33, 40, 39, 36, 35,      // erb1 g1s,g1b,c1w,c1b,g2s,g2b,c2w,c2b
  46, 45, 42, 41, 48, 47, 44, 43,      // erb2
  32, 31, 2, 1,                        // e4w,e4b,d1w,d1b
  8, 7, 4, 3, 10, 9, 6, 5,             // drb1
  16, 15, 12, 11, 18, 17, 14, 13,      // drb2
  20, 19, 22, 21, 24, 23,              // dt1..dt3
  50, 49, 52, 51, 54, 53,              // s1..s3
  0                                    // codebook
};

extern "C" void kernel_launch(void* const* d_in, const int* in_sizes, int n_in,
                              void* d_out, int out_size, void* d_ws, size_t ws_size,
                              hipStream_t stream)
{
  if (n_in < 56) return;
  if (ws_size < (size_t)WS_FLOATS * sizeof(float)) return;
  float* ws = (float*)d_ws;
  float* out = (float*)d_out;

  const int* tab = (in_sizes[0] == 3145728) ? INS : SRT;
  const float* P[N_T];
  for (int k = 0; k < N_T; ++k) P[k] = (const float*)d_in[tab[k]];

  // ---- encoder ----
  k_conv_e1<<<(8388608 + 255) / 256, 256, 0, stream>>>(P[iX], P[iE1W], P[iE1B], ws + WA);
  convl(stream, ws + WA, P[iE2W], P[iE2B], ws + WB, 16, 32, 128, 128, 64, 64, 64, 4, 2, 1, 1, 1);
  convl(stream, ws + WB, P[iE3W], P[iE3B], ws + WT2, 16, 64, 64, 64, 128, 32, 32, 4, 2, 1, 1, 1);
  resblock(stream, ws + WT2, ws + WT3, ws + WT4,
           P[iERB1_G1S], P[iERB1_G1B], P[iERB1_C1W], P[iERB1_C1B],
           P[iERB1_G2S], P[iERB1_G2B], P[iERB1_C2W], P[iERB1_C2B]);
  resblock(stream, ws + WT2, ws + WT3, ws + WT4,
           P[iERB2_G1S], P[iERB2_G1B], P[iERB2_C1W], P[iERB2_C1B],
           P[iERB2_G2S], P[iERB2_G2B], P[iERB2_C2W], P[iERB2_C2B]);
  convl(stream, ws + WT2, P[iE4W], P[iE4B], ws + WT0, 16, 128, 32, 32, 128, 32, 32, 1, 1, 0, 1, 0);

  // ---- vector quantizer ----
  k_codenorm<<<4, 256, 0, stream>>>(P[iCODEBOOK], ws + WCN);
  k_vq<<<1024, 32, 0, stream>>>(ws + WT0, P[iCODEBOOK], ws + WCN,
                                ws + WT1, out + IDX_OFF, ws + WLB);
  k_vq_fin<<<1, 1, 0, stream>>>(ws + WLB, out + VQLOSS_OFF);

  // ---- decoder ----
  convl(stream, ws + WT1, P[iD1W], P[iD1B], ws + WT2, 16, 128, 32, 32, 128, 32, 32, 1, 1, 0, 1, 0);
  resblock(stream, ws + WT2, ws + WT3, ws + WT4,
           P[iDRB1_G1S], P[iDRB1_G1B], P[iDRB1_C1W], P[iDRB1_C1B],
           P[iDRB1_G2S], P[iDRB1_G2B], P[iDRB1_C2W], P[iDRB1_C2B]);
  resblock(stream, ws + WT2, ws + WT3, ws + WT4,
           P[iDRB2_G1S], P[iDRB2_G1B], P[iDRB2_C1W], P[iDRB2_C1B],
           P[iDRB2_G2S], P[iDRB2_G2B], P[iDRB2_C2W], P[iDRB2_C2B]);
  // transposed convs as lhs-dilated convs (stride 1, pad 2, dil 2)
  convl(stream, ws + WT2, P[iDT1W], P[iDT1B], ws + WB, 16, 128, 32, 32, 64, 64, 64, 4, 1, 2, 2, 1);
  convl(stream, ws + WB, P[iDT2W], P[iDT2B], ws + WA, 16, 64, 64, 64, 32, 128, 128, 4, 1, 2, 2, 1);
  convl(stream, ws + WA, P[iDT3W], P[iDT3B], out + RECON_OFF, 16, 32, 128, 128, 3, 256, 256, 4, 1, 2, 2, 2);

  // ---- segmentation head ----
  convl(stream, ws + WT1, P[iS1W], P[iS1B], ws + WT3, 16, 128, 32, 32, 64, 32, 32, 3, 1, 1, 1, 1);
  k_up2<<<(4194304 + 255) / 256, 256, 0, stream>>>(ws + WT3, ws + WB, 64, 32, 32);
  convl(stream, ws + WB, P[iS2W], P[iS2B], ws + WT4, 16, 64, 64, 64, 32, 64, 64, 3, 1, 1, 1, 1);
  k_up2<<<(8388608 + 255) / 256, 256, 0, stream>>>(ws + WT4, ws + WA, 32, 64, 64);
  convl(stream, ws + WA, P[iS3W], P[iS3B], ws + WC, 16, 32, 128, 128, 19, 128, 128, 3, 1, 1, 1, 0);
  // final up2 to 256x256; trailing resize(256->256) is identity
  k_up2<<<(19922944 + 255) / 256, 256, 0, stream>>>(ws + WC, out + SEG_OFF, 19, 128, 128);
}
